// LSTMAttentionModel_44564580663602
// MI455X (gfx1250) — compile-verified, matrix operand order checked by reading, not run on hardware
//
#include <hip/hip_runtime.h>
#include <hip/hip_bf16.h>

#define T_SEQ 200
#define B_SZ  1024
#define E_DIM 128
#define H_DIM 128
#define V_SZ  100000
#define G4H   512
#define TP    224      // padded seq: 14*16, 7*32
#define NT_T  14       // TP/16
#define NC_T  7        // TP/32
#define NT_H  8        // H/16

typedef __bf16 bf16;
typedef __attribute__((ext_vector_type(16))) __bf16 v16bf;
typedef __attribute__((ext_vector_type(8)))  float  v8f;

static __device__ inline v8f wmma_bf16(v16bf a, v16bf b, v8f c) {
  // (neg_a, A, neg_b, B, c_mod, C, reuse_a, reuse_b)
  return __builtin_amdgcn_wmma_f32_16x16x32_bf16(false, a, false, b, (short)0, c, false, false);
}

// Position of element (row, kp) inside a 512-element 16-bit fragment.
// ISA 7.12.2: lane = half*16 + row; K interleave: elems 0..7 -> K 0..7 (half0),
// elems 8..15 -> K 16..23 (half0); +8 for half1.
static __device__ inline int frag_off(int row, int kp) {
  int half = (kp >> 3) & 1;
  int i = (kp & 7) | ((kp & 16) >> 1);
  return (half * 16 + row) * 16 + i;
}
static __device__ inline int kmap(int half, int i) {
  return (i < 8 ? i : i + 8) + half * 8;
}

// ---------------- conversion / packing ----------------

__global__ void k_cvt(const float* __restrict__ src, bf16* __restrict__ dst, long n) {
  long stride = (long)gridDim.x * blockDim.x;
  for (long i = (long)blockIdx.x * blockDim.x + threadIdx.x; i < n; i += stride)
    dst[i] = (bf16)src[i];
}

// Pack W [N,K] row-major fp32 into B-fragment bf16 (B[k][n] = W[n][k]).
__global__ void k_pack_b(const float* __restrict__ w, bf16* __restrict__ dst, int N, int K) {
  long total = (long)N * K;
  long kc = K >> 5;
  long stride = (long)gridDim.x * blockDim.x;
  for (long idx = (long)blockIdx.x * blockDim.x + threadIdx.x; idx < total; idx += stride) {
    int nn = (int)(idx / K), k = (int)(idx % K);
    long frag = (long)(nn >> 4) * kc + (k >> 5);
    dst[frag * 512 + frag_off(nn & 15, k & 31)] = (bf16)w[idx];
  }
}

// ---------------- LSTM: 64 blocks x 16 batches, 200 serial steps ----------------

__global__ __launch_bounds__(128) void k_lstm(
    const int* __restrict__ x, const int* __restrict__ lengths,
    const bf16* __restrict__ embRows,
    const bf16* __restrict__ wihFr, const bf16* __restrict__ whhFr,
    const float* __restrict__ b_ih, const float* __restrict__ b_hh,
    bf16* __restrict__ lstmFr) {
  __shared__ float sZ[16][G4H];                 // 32 KB
  __shared__ float sC[16][H_DIM];               // 8 KB
  __shared__ __align__(32) bf16 sH[4 * 512];    // h in A-frag order, 4 KB
  __shared__ __align__(32) bf16 sX[4 * 512];    // x_t in A-frag order, 4 KB
  __shared__ int sTok[16];
  __shared__ int sLen[16];

  int tid = threadIdx.x, lane = tid & 31, wv = tid >> 5;
  int b0 = blockIdx.x * 16;
  int nloc = lane & 15, mh = lane >> 4;

  for (int j = tid; j < 16 * H_DIM; j += 128) sC[j >> 7][j & 127] = 0.f;
  for (int j = tid; j < 2048; j += 128) sH[j] = (bf16)0.f;
  if (tid < 16) sLen[tid] = lengths[b0 + tid];
  __syncthreads();

  for (int t = 0; t < T_SEQ; ++t) {
    if (tid < 16) sTok[tid] = x[t * B_SZ + b0 + tid];
    __syncthreads();
    for (int j = tid; j < 2048; j += 128) {        // gather embeddings -> A-frags
      int row = j >> 7, kp = j & 127;
      float v = (float)embRows[(long)sTok[row] * E_DIM + kp];
      sX[(kp >> 5) * 512 + frag_off(row, kp & 31)] = (bf16)v;
    }
    __syncthreads();

    // z[16,512] = x@W_ih^T + h@W_hh^T : each wave owns 8 of 32 n-tiles.
    // unroll 2 keeps VGPRs < 256 (avoids s_set_vgpr_msb churn, keeps occupancy).
#pragma unroll 2
    for (int nti = 0; nti < 8; ++nti) {
      int nt = wv * 8 + nti;
      v8f acc = {};
#pragma unroll
      for (int c = 0; c < 4; ++c) {
        v16bf a  = *(const v16bf*)(sX + c * 512 + lane * 16);
        v16bf bb = *(const v16bf*)(wihFr + ((long)nt * 4 + c) * 512 + lane * 16);
        acc = wmma_bf16(a, bb, acc);
      }
#pragma unroll
      for (int c = 0; c < 4; ++c) {
        v16bf a  = *(const v16bf*)(sH + c * 512 + lane * 16);
        v16bf bb = *(const v16bf*)(whhFr + ((long)nt * 4 + c) * 512 + lane * 16);
        acc = wmma_bf16(a, bb, acc);
      }
      union { v8f v; float f[8]; } u; u.v = acc;
#pragma unroll
      for (int r = 0; r < 8; ++r) sZ[r + 8 * mh][nt * 16 + nloc] = u.f[r];
    }
    __syncthreads();

    // fused gates
    for (int j = tid; j < 2048; j += 128) {
      int row = j >> 7, hj = j & 127;
      float zi = sZ[row][hj]       + b_ih[hj]       + b_hh[hj];
      float zf = sZ[row][128 + hj] + b_ih[128 + hj] + b_hh[128 + hj];
      float zg = sZ[row][256 + hj] + b_ih[256 + hj] + b_hh[256 + hj];
      float zo = sZ[row][384 + hj] + b_ih[384 + hj] + b_hh[384 + hj];
      float ig = 1.f / (1.f + __expf(-zi));
      float fg = 1.f / (1.f + __expf(-zf));
      float gg = tanhf(zg);
      float og = 1.f / (1.f + __expf(-zo));
      float cc = fg * sC[row][hj] + ig * gg;
      sC[row][hj] = cc;
      float h = og * tanhf(cc);
      float hm = (t < sLen[row]) ? h : 0.f;            // mask only the output
      long tile = (long)(b0 + row) * NT_T + (t >> 4);  // A-frag packed lstm_out
      lstmFr[(tile * 4 + (hj >> 5)) * 512 + frag_off(t & 15, hj & 31)] = (bf16)hm;
      sH[(hj >> 5) * 512 + frag_off(row, hj & 31)] = (bf16)h;
    }
    __syncthreads();
  }
}

// ---------------- fused QKV: one wave per 16-row tile, reuses A frags ----------------

__global__ __launch_bounds__(128) void k_qkv(
    const bf16* __restrict__ lstmFr,
    const bf16* __restrict__ wqFr, const bf16* __restrict__ wkFr, const bf16* __restrict__ wvFr,
    const float* __restrict__ bq, const float* __restrict__ bk, const float* __restrict__ bv,
    bf16* __restrict__ qFr, bf16* __restrict__ kFr, bf16* __restrict__ vFr) {
  int lane = threadIdx.x & 31, wv = threadIdx.x >> 5;
  long m_tile = (long)blockIdx.x * 4 + wv;          // B*NT_T tiles
  int b = (int)(m_tile / NT_T), ttile = (int)(m_tile % NT_T);
  int nloc = lane & 15, mh = lane >> 4;

  v16bf a[4];
#pragma unroll
  for (int c = 0; c < 4; ++c)
    a[c] = *(const v16bf*)(lstmFr + (m_tile * 4 + c) * 512 + lane * 16);

#pragma unroll 2
  for (int nt = 0; nt < NT_H; ++nt) {
    v8f aq = {}, ak = {}, av = {};
#pragma unroll
    for (int c = 0; c < 4; ++c) {
      v16bf bqf = *(const v16bf*)(wqFr + ((long)nt * 4 + c) * 512 + lane * 16);
      v16bf bkf = *(const v16bf*)(wkFr + ((long)nt * 4 + c) * 512 + lane * 16);
      v16bf bvf = *(const v16bf*)(wvFr + ((long)nt * 4 + c) * 512 + lane * 16);
      aq = wmma_bf16(a[c], bqf, aq);
      ak = wmma_bf16(a[c], bkf, ak);
      av = wmma_bf16(a[c], bvf, av);
    }
    float biq = bq[nt * 16 + nloc], bik = bk[nt * 16 + nloc], biv = bv[nt * 16 + nloc];
    union { v8f v; float f[8]; } uq, uk, uv; uq.v = aq; uk.v = ak; uv.v = av;
#pragma unroll
    for (int r = 0; r < 8; ++r) {
      int Ml = r + 8 * mh;
      int t = ttile * 16 + Ml;
      int hh = nt * 16 + nloc;
      // q as A-fragments (rows = t)
      qFr[(m_tile * 4 + (hh >> 5)) * 512 + frag_off(Ml, hh & 31)] = (bf16)(uq.f[r] + biq);
      // k as B-fragments for scores: k-dim = h, n-dim = t  (14 ntiles x 4 chunks)
      kFr[((long)b * 56 + ttile * 4 + (hh >> 5)) * 512 + frag_off(Ml, hh & 31)] = (bf16)(uk.f[r] + bik);
      // v as B-fragments for P@V: k-dim = t, n-dim = h     (8 ntiles x 7 chunks)
      vFr[((long)b * 56 + nt * 7 + (t >> 5)) * 512 + frag_off(nloc, t & 31)] = (bf16)(uv.f[r] + biv);
    }
  }
}

// ---------------- attention + pooling: one block (2 waves) per batch ----------------

__global__ __launch_bounds__(64) void k_attn(
    const bf16* __restrict__ qFr, const bf16* __restrict__ kFr, const bf16* __restrict__ vFr,
    const int* __restrict__ lengths, float* __restrict__ pooled) {
  __shared__ float sS[2][16][TP];                       // 28 KB
  __shared__ __align__(32) bf16 sP[2][NC_T * 512];      // 14 KB
  __shared__ float sPool[H_DIM];
  int tid = threadIdx.x, lane = tid & 31, wv = tid >> 5;
  int b = blockIdx.x;
  int nloc = lane & 15, mh = lane >> 4;
  for (int j = tid; j < H_DIM; j += 64) sPool[j] = 0.f;
  __syncthreads();
  int len = lengths[b];
  const float rsq = 0.08838834764831845f;               // 1/sqrt(128)

  for (int qt = wv; qt < 13; qt += 2) {                 // 13 q-tiles cover t<208
    v16bf qa[4];
#pragma unroll
    for (int c = 0; c < 4; ++c)
      qa[c] = *(const v16bf*)(qFr + (((long)b * NT_T + qt) * 4 + c) * 512 + lane * 16);

#pragma unroll 2
    for (int nt = 0; nt < NT_T; ++nt) {                 // scores tile row
      v8f acc = {};
#pragma unroll
      for (int c = 0; c < 4; ++c) {
        v16bf kb = *(const v16bf*)(kFr + ((long)b * 56 + nt * 4 + c) * 512 + lane * 16);
        acc = wmma_bf16(qa[c], kb, acc);
      }
      int kj = nt * 16 + nloc;
      union { v8f v; float f[8]; } u; u.v = acc;
#pragma unroll
      for (int r = 0; r < 8; ++r) {
        float s = u.f[r] * rsq;
        if (kj >= len) s = -1e30f;                      // mask invalid + pad keys
        sS[wv][r + 8 * mh][kj] = s;
      }
    }
    __asm__ volatile("s_wait_dscnt 0" ::: "memory");    // wave-local LDS visibility

    if (lane < 16) {                                    // row softmax -> P frags (bf16)
      int row = lane;
      float m = -1e30f;
      for (int k = 0; k < TP; ++k) m = fmaxf(m, sS[wv][row][k]);
      float sum = 0.f;
      for (int k = 0; k < TP; ++k) sum += __expf(sS[wv][row][k] - m);
      float inv = 1.f / sum;
      for (int k = 0; k < TP; ++k) {
        float p = __expf(sS[wv][row][k] - m) * inv;
        sP[wv][(k >> 5) * 512 + frag_off(row, k & 31)] = (bf16)p;
      }
    }
    __asm__ volatile("s_wait_dscnt 0" ::: "memory");

#pragma unroll 2
    for (int ht = 0; ht < NT_H; ++ht) {                 // weighted = P @ V, pool
      v8f acc = {};
#pragma unroll
      for (int tc = 0; tc < NC_T; ++tc) {
        v16bf pa = *(const v16bf*)(sP[wv] + tc * 512 + lane * 16);
        v16bf vb = *(const v16bf*)(vFr + ((long)b * 56 + ht * 7 + tc) * 512 + lane * 16);
        acc = wmma_bf16(pa, vb, acc);
      }
      union { v8f v; float f[8]; } u; u.v = acc;
#pragma unroll
      for (int r = 0; r < 8; ++r) {
        int qrow = qt * 16 + r + 8 * mh;
        if (qrow < T_SEQ) atomicAdd(&sPool[ht * 16 + nloc], u.f[r]);
      }
    }
  }
  __syncthreads();
  for (int j = tid; j < H_DIM; j += 64)
    pooled[(long)b * H_DIM + j] = sPool[j] * (1.f / T_SEQ);
}

// ---------------- feat = pooled @ Wo^T + bo (stored as A-frags) ----------------

__global__ __launch_bounds__(128) void k_feat(
    const float* __restrict__ pooled, const bf16* __restrict__ woFr,
    const float* __restrict__ bo, bf16* __restrict__ featFr) {
  int lane = threadIdx.x & 31, wv = threadIdx.x >> 5;
  int m_tile = blockIdx.x * 4 + wv;                     // 64 tiles of B
  int nloc = lane & 15, mh = lane >> 4;
  v16bf a[4];
#pragma unroll
  for (int c = 0; c < 4; ++c) {
    union { v16bf v; bf16 e[16]; } u;
#pragma unroll
    for (int i = 0; i < 16; ++i)
      u.e[i] = (bf16)pooled[(long)(m_tile * 16 + nloc) * H_DIM + c * 32 + kmap(mh, i)];
    a[c] = u.v;
  }
#pragma unroll 2
  for (int nt = 0; nt < NT_H; ++nt) {
    v8f acc = {};
#pragma unroll
    for (int c = 0; c < 4; ++c) {
      v16bf bb = *(const v16bf*)(woFr + ((long)nt * 4 + c) * 512 + lane * 16);
      acc = wmma_bf16(a[c], bb, acc);
    }
    float bias = bo[nt * 16 + nloc];
    union { v8f v; float f[8]; } u; u.v = acc;
#pragma unroll
    for (int r = 0; r < 8; ++r) {
      int Ml = r + 8 * mh;
      int kk = nt * 16 + nloc;
      featFr[((long)m_tile * 4 + (kk >> 5)) * 512 + frag_off(Ml, kk & 31)] = (bf16)(u.f[r] + bias);
    }
  }
}

// ---------------- scores = feat @ emb^T : HBM-write-bound vocab GEMM ----------------

__global__ __launch_bounds__(256) void k_scores(
    const bf16* __restrict__ featFr, const bf16* __restrict__ embFr, float* __restrict__ out) {
  int lane = threadIdx.x & 31, wv = threadIdx.x >> 5;
  int mt = blockIdx.x;                                  // 64
  long nt = (long)blockIdx.y * 8 + wv;                  // 6250 vocab n-tiles
  if (nt >= V_SZ / 16) return;
  v16bf a[4];
#pragma unroll
  for (int c = 0; c < 4; ++c)
    a[c] = *(const v16bf*)(featFr + ((long)mt * 4 + c) * 512 + lane * 16);
  v8f acc = {};
#pragma unroll
  for (int c = 0; c < 4; ++c) {
    v16bf bb = *(const v16bf*)(embFr + (nt * 4 + c) * 512 + lane * 16);
    acc = wmma_bf16(a[c], bb, acc);
  }
  int nloc = lane & 15, mh = lane >> 4;
  union { v8f v; float f[8]; } u; u.v = acc;
#pragma unroll
  for (int r = 0; r < 8; ++r) {
    long row = (long)mt * 16 + r + 8 * mh;
    out[row * V_SZ + nt * 16 + nloc] = u.f[r];
  }
}

// ---------------- host launcher ----------------

extern "C" void kernel_launch(void* const* d_in, const int* in_sizes, int n_in,
                              void* d_out, int out_size, void* d_ws, size_t ws_size,
                              hipStream_t stream) {
  (void)in_sizes; (void)n_in; (void)out_size; (void)ws_size;
  const int*   x    = (const int*)d_in[0];
  const int*   len  = (const int*)d_in[1];
  const float* emb  = (const float*)d_in[2];
  const float* W_ih = (const float*)d_in[3];
  const float* W_hh = (const float*)d_in[4];
  const float* b_ih = (const float*)d_in[5];
  const float* b_hh = (const float*)d_in[6];
  const float* Wq   = (const float*)d_in[7];
  const float* bq   = (const float*)d_in[8];
  const float* Wk   = (const float*)d_in[9];
  const float* bk   = (const float*)d_in[10];
  const float* Wv   = (const float*)d_in[11];
  const float* bv   = (const float*)d_in[12];
  const float* Wo   = (const float*)d_in[13];
  const float* bo   = (const float*)d_in[14];
  float* out = (float*)d_out;

  char* ws = (char*)d_ws;
  size_t off = 0;
  auto alloc = [&](size_t bytes) -> void* {
    void* p = ws + off;
    off = (off + bytes + 255) & ~(size_t)255;
    return p;
  };
  const size_t fragBytes = 512 * sizeof(bf16);
  bf16* embRows = (bf16*)alloc((size_t)V_SZ * E_DIM * 2);
  bf16* embFr   = (bf16*)alloc((size_t)(V_SZ / 16) * 4 * fragBytes);
  bf16* wihFr   = (bf16*)alloc((size_t)(G4H / 16) * 4 * fragBytes);
  bf16* whhFr   = (bf16*)alloc((size_t)(G4H / 16) * 4 * fragBytes);
  bf16* wqFr    = (bf16*)alloc((size_t)NT_H * 4 * fragBytes);
  bf16* wkFr    = (bf16*)alloc((size_t)NT_H * 4 * fragBytes);
  bf16* wvFr    = (bf16*)alloc((size_t)NT_H * 4 * fragBytes);
  bf16* woFr    = (bf16*)alloc((size_t)NT_H * 4 * fragBytes);
  size_t lstmBytes = (size_t)B_SZ * NT_T * 4 * fragBytes;
  bf16* lstmFr  = (bf16*)alloc(lstmBytes);
  bf16* qFr     = (bf16*)alloc(lstmBytes);
  bf16* kFr     = (bf16*)alloc((size_t)B_SZ * 56 * fragBytes);
  bf16* vFr     = (bf16*)alloc((size_t)B_SZ * 56 * fragBytes);
  float* pooled = (float*)alloc((size_t)B_SZ * H_DIM * sizeof(float));
  bf16* featFr  = (bf16*)alloc((size_t)(B_SZ / 16) * 4 * fragBytes);

  hipMemsetAsync(lstmFr, 0, lstmBytes, stream);   // zero pad rows t in [200,224)

  k_cvt   <<<512, 256, 0, stream>>>(emb, embRows, (long)V_SZ * E_DIM);
  k_pack_b<<<512, 256, 0, stream>>>(emb,  embFr, V_SZ, E_DIM);
  k_pack_b<<< 64, 256, 0, stream>>>(W_ih, wihFr, G4H,  E_DIM);
  k_pack_b<<< 64, 256, 0, stream>>>(W_hh, whhFr, G4H,  H_DIM);
  k_pack_b<<< 16, 256, 0, stream>>>(Wq,   wqFr,  H_DIM, H_DIM);
  k_pack_b<<< 16, 256, 0, stream>>>(Wk,   wkFr,  H_DIM, H_DIM);
  k_pack_b<<< 16, 256, 0, stream>>>(Wv,   wvFr,  H_DIM, H_DIM);
  k_pack_b<<< 16, 256, 0, stream>>>(Wo,   woFr,  H_DIM, H_DIM);

  k_lstm<<<B_SZ / 16, 128, 0, stream>>>(x, len, embRows, wihFr, whhFr, b_ih, b_hh, lstmFr);
  k_qkv <<<B_SZ * NT_T / 4, 128, 0, stream>>>(lstmFr, wqFr, wkFr, wvFr, bq, bk, bv, qFr, kFr, vFr);
  k_attn<<<B_SZ, 64, 0, stream>>>(qFr, kFr, vFr, len, pooled);
  k_feat<<<B_SZ / 16 / 4, 128, 0, stream>>>(pooled, woFr, bo, featFr);
  dim3 g(B_SZ / 16, (V_SZ / 16 + 7) / 8);
  k_scores<<<g, 256, 0, stream>>>(featFr, embFr, out);
}
